// MHALayerModule_4466765988515
// MI455X (gfx1250) — compile-verified, matrix operand order checked by reading, not run on hardware
//
#include <hip/hip_runtime.h>
#include <hip/hip_bf16.h>

// ---------------------------------------------------------------------------
// MHALayerModule for MI455X (gfx1250): heavy 1x1 convs as f16 WMMA GEMMs
// (v_wmma_f32_16x16x32_f16) with 64x32 per-wave register blocking (spill-free),
// everything else bandwidth-bound VALU kernels.
// ---------------------------------------------------------------------------

typedef __attribute__((ext_vector_type(16))) _Float16 v16h;
typedef __attribute__((ext_vector_type(8)))  _Float16 v8h;
typedef __attribute__((ext_vector_type(8)))  float    v8f;

#define BATCH 8
#define CDIM  512
#define RES   56
#define HW    3136      // 56*56
#define HW2   7
#define NTOK  49
#define HEADS 8
#define HD    64
#define HIDC  2048
#define GEPS  1e-5f

__device__ __forceinline__ float hswish(float x) {
    float t = fminf(fmaxf(x + 3.0f, 0.0f), 6.0f);
    return x * t * (1.0f / 6.0f);
}
__device__ __forceinline__ float bnval(float x, float g, float be, float m, float v) {
    return (x - m) * g * rsqrtf(v + GEPS) + be;
}

// --------------------------- f32 -> f16 convert -----------------------------
__global__ void k_cvt(const float* __restrict__ in, _Float16* __restrict__ out, size_t n) {
    size_t i = (size_t)blockIdx.x * blockDim.x + threadIdx.x;
    if (i < n) out[i] = (_Float16)in[i];
}

// --------------------------- depthwise 3x3 on 56x56 -------------------------
// out = (addIdentity ? in : 0) + dwconv3x3(in), pad=1. Layout NCHW flat.
__global__ void k_dwconv(const float* __restrict__ in, const float* __restrict__ w,
                         float* __restrict__ out, int Ctot, int addIdentity, size_t total) {
    size_t i = (size_t)blockIdx.x * blockDim.x + threadIdx.x;
    if (i >= total) return;
    int p = (int)(i % HW);
    int c = (int)((i / HW) % Ctot);
    const float* plane = in + (i - p);
    const float* wc = w + (size_t)c * 9;
    int y = p / RES, x = p % RES;
    float s = 0.0f;
#pragma unroll
    for (int ky = 0; ky < 3; ++ky) {
        int iy = y + ky - 1;
        if (iy < 0 || iy >= RES) continue;
#pragma unroll
        for (int kx = 0; kx < 3; ++kx) {
            int ix = x + kx - 1;
            if (ix < 0 || ix >= RES) continue;
            s += plane[iy * RES + ix] * wc[ky * 3 + kx];
        }
    }
    out[i] = (addIdentity ? in[i] : 0.0f) + s;
}

// ------------------ stride-8 depthwise conv + bias + BN -> f16 tokens -------
__global__ void k_stride_bn(const float* __restrict__ xl, const float* __restrict__ scw,
                            const float* __restrict__ scb, const float* __restrict__ ls1,
                            const float* __restrict__ g, const float* __restrict__ be,
                            const float* __restrict__ mm, const float* __restrict__ vv,
                            _Float16* __restrict__ xsch) {
    int i = blockIdx.x * blockDim.x + threadIdx.x;
    if (i >= BATCH * CDIM * NTOK) return;
    int n = i % NTOK;
    int c = (i / NTOK) % CDIM;
    int b = i / (NTOK * CDIM);
    int oy = n / HW2, ox = n % HW2;
    const float* plane = xl + ((size_t)b * CDIM + c) * HW;
    const float* wc = scw + (size_t)c * 9;
    float s = 0.0f;
#pragma unroll
    for (int ky = 0; ky < 3; ++ky) {
        int iy = oy * 8 + ky - 1;
        if (iy < 0 || iy >= RES) continue;
#pragma unroll
        for (int kx = 0; kx < 3; ++kx) {
            int ix = ox * 8 + kx - 1;
            if (ix < 0 || ix >= RES) continue;
            s += plane[iy * RES + ix] * wc[ky * 3 + kx];
        }
    }
    s = s * ls1[c] + scb[c];
    float o = bnval(s, g[c], be[c], mm[c], vv[c]);
    xsch[((size_t)b * NTOK + n) * CDIM + c] = (_Float16)o;
}

// ------------------------------ WMMA GEMM -----------------------------------
// out[b][m][p] = sum_k W[m][k] * Bt[b][p][k], optional BN epilogue per row m.
// One wave per 64x32 output tile: 4 A frags x 2 B frags -> 8 WMMAs per K-step
// of 32 (12 b128 loads per 8 WMMAs; A reused 2x, B reused 4x). ~130 VGPRs:
// 8 f32 accumulators (64) + 6 fragments (48) + addressing, so no spills.
// Fragment loads follow the ISA 16-bit A/B lane layout: lane half `hi` owns
// K runs [hi*8, hi*8+8) and [16+hi*8, 16+hi*8+8).
// All control flow is wave-uniform so EXEC stays all-ones for WMMA.
__global__ void __launch_bounds__(128, 1)
k_gemm_bn(const _Float16* __restrict__ W, const _Float16* __restrict__ Bt,
          float* __restrict__ out, int M, int K, int P,
          const float* g, const float* be, const float* mm, const float* vv) {
    const int lane = threadIdx.x & 31;
    const int wave = threadIdx.x >> 5;
    const int nTilesN = (P + 31) >> 5;
    const int nTiles = (M >> 6) * nTilesN;
    int tile = blockIdx.x * (blockDim.x >> 5) + wave;
    if (tile >= nTiles) return;                 // wave-uniform exit
    const int tm = (tile / nTilesN) << 6;
    const int tn = (tile % nTilesN) << 5;
    const int b = blockIdx.y;
    const int hi = lane >> 4;
    const int lo = lane & 15;

    const _Float16* wrow[4];
    const _Float16* brow[2];
    int cols[2];
#pragma unroll
    for (int t = 0; t < 4; ++t)
        wrow[t] = W + (size_t)(tm + t * 16 + lo) * K;
#pragma unroll
    for (int t = 0; t < 2; ++t) {
        int col = tn + t * 16 + lo;
        cols[t] = col;
        int colc = col < P ? col : P - 1;
        brow[t] = Bt + ((size_t)b * P + colc) * K;
    }

    v8f acc[8] = {};
    union U { v16h v; v8h h[2]; };
    for (int k0 = 0; k0 < K; k0 += 32) {
        U a[4], bf[2];
#pragma unroll
        for (int t = 0; t < 4; ++t) {
            a[t].h[0] = *(const v8h*)(wrow[t] + k0 + hi * 8);
            a[t].h[1] = *(const v8h*)(wrow[t] + k0 + 16 + hi * 8);
        }
#pragma unroll
        for (int t = 0; t < 2; ++t) {
            bf[t].h[0] = *(const v8h*)(brow[t] + k0 + hi * 8);
            bf[t].h[1] = *(const v8h*)(brow[t] + k0 + 16 + hi * 8);
        }
#pragma unroll
        for (int mi = 0; mi < 4; ++mi)
#pragma unroll
            for (int ni = 0; ni < 2; ++ni)
                acc[mi * 2 + ni] = __builtin_amdgcn_wmma_f32_16x16x32_f16(
                    false, a[mi].v, false, bf[ni].v, (short)0, acc[mi * 2 + ni],
                    false, false);
    }
#pragma unroll
    for (int mi = 0; mi < 4; ++mi) {
#pragma unroll
        for (int ni = 0; ni < 2; ++ni) {
            int col = cols[ni];
            bool ok = col < P;
#pragma unroll
            for (int r = 0; r < 8; ++r) {
                int m = tm + mi * 16 + r + hi * 8;
                float val = acc[mi * 2 + ni][r];
                if (g) val = bnval(val, g[m], be[m], mm[m], vv[m]);
                if (ok) out[((size_t)b * M + m) * P + col] = val;
            }
        }
    }
}

// ---------------------- local 3x3 grouped conv on 7x7 (v_local) -------------
__global__ void k_vlocal(const float* __restrict__ v4, const float* __restrict__ lw,
                         const float* __restrict__ lb, const float* __restrict__ g,
                         const float* __restrict__ be, const float* __restrict__ mm,
                         const float* __restrict__ vv, float* __restrict__ vloc) {
    int i = blockIdx.x * blockDim.x + threadIdx.x;
    if (i >= BATCH * CDIM * NTOK) return;
    int n = i % NTOK;
    int co = (i / NTOK) % CDIM;
    int b = i / (NTOK * CDIM);
    int grp = co >> 6;
    int oy = n / HW2, ox = n % HW2;
    float s = 0.0f;
    for (int ci = 0; ci < HD; ++ci) {
        const float* vp = v4 + ((size_t)b * CDIM + grp * HD + ci) * NTOK;
        const float* wp = lw + ((size_t)co * HD + ci) * 9;
#pragma unroll
        for (int ky = 0; ky < 3; ++ky) {
            int iy = oy + ky - 1;
            if (iy < 0 || iy >= HW2) continue;
#pragma unroll
            for (int kx = 0; kx < 3; ++kx) {
                int ix = ox + kx - 1;
                if (ix < 0 || ix >= HW2) continue;
                s += vp[iy * HW2 + ix] * wp[ky * 3 + kx];
            }
        }
    }
    s += lb[co];
    vloc[i] = bnval(s, g[co], be[co], mm[co], vv[co]);
}

// ----------------------------- attention norms ------------------------------
__global__ void k_qnorm(const float* __restrict__ q, float* __restrict__ qn) {
    int t = blockIdx.x * blockDim.x + threadIdx.x;       // (b,h,n)
    if (t >= BATCH * HEADS * NTOK) return;
    int n = t % NTOK;
    int h = (t / NTOK) % HEADS;
    int b = t / (NTOK * HEADS);
    const float* qp = q + ((size_t)b * CDIM + h * HD) * NTOK + n;
    float s = 0.0f;
    for (int d = 0; d < HD; ++d) { float v = qp[d * NTOK]; s += v * v; }
    qn[t] = sqrtf(s);
}
__global__ void k_knorm(const float* __restrict__ k, float* __restrict__ kn) {
    int t = blockIdx.x * blockDim.x + threadIdx.x;       // (b,h,d) == (b,c)
    if (t >= BATCH * CDIM) return;
    const float* kp = k + (size_t)t * NTOK;
    float s = 0.0f;
    for (int m = 0; m < NTOK; ++m) { float v = kp[m]; s += v * v; }
    kn[t] = sqrtf(s);
}

// ------------------------------ logits --------------------------------------
__global__ void k_logits(const float* __restrict__ q, const float* __restrict__ k,
                         const float* __restrict__ qn, const float* __restrict__ kn,
                         const float* __restrict__ lsr, const float* __restrict__ abias,
                         const int* __restrict__ idxs, float* __restrict__ att) {
    int i = blockIdx.x * blockDim.x + threadIdx.x;       // (b,h,n,m)
    if (i >= BATCH * HEADS * NTOK * NTOK) return;
    int m = i % NTOK;
    int n = (i / NTOK) % NTOK;
    int h = (i / (NTOK * NTOK)) % HEADS;
    int b = i / (NTOK * NTOK * HEADS);
    const float* qp = q + ((size_t)b * CDIM + h * HD) * NTOK + n;
    const float* kp = k + ((size_t)b * CDIM + h * HD) * NTOK + m;
    const float* knp = kn + (b * HEADS + h) * HD;
    float acc = 0.0f;
    for (int d = 0; d < HD; ++d)
        acc += qp[d * NTOK] * kp[d * NTOK] / fmaxf(knp[d], 1e-12f);
    float qq = fmaxf(qn[(b * HEADS + h) * NTOK + n], 1e-12f);
    float ls = __expf(fminf(lsr[h], 4.6051701859880914f)); // ln(100)
    att[i] = acc / qq * ls + abias[h * NTOK + idxs[n * NTOK + m]];
}

// --------------------------- talking-head mix -------------------------------
__global__ void k_thmix(const float* __restrict__ in, const float* __restrict__ thw,
                        float* __restrict__ out) {
    int i = blockIdx.x * blockDim.x + threadIdx.x;       // (b,h,n,m)
    if (i >= BATCH * HEADS * NTOK * NTOK) return;
    int nm = i % (NTOK * NTOK);
    int h = (i / (NTOK * NTOK)) % HEADS;
    int b = i / (NTOK * NTOK * HEADS);
    float s = 0.0f;
#pragma unroll
    for (int gg = 0; gg < HEADS; ++gg)
        s += thw[h * HEADS + gg] * in[((size_t)b * HEADS + gg) * (NTOK * NTOK) + nm];
    out[i] = s;
}

// ------------------------------ softmax rows --------------------------------
__global__ void k_softmax(float* __restrict__ att) {
    int t = blockIdx.x * blockDim.x + threadIdx.x;       // (b,h,n)
    if (t >= BATCH * HEADS * NTOK) return;
    float* row = att + (size_t)t * NTOK;
    float mx = -1e30f;
    for (int m = 0; m < NTOK; ++m) mx = fmaxf(mx, row[m]);
    float sum = 0.0f;
    for (int m = 0; m < NTOK; ++m) { float e = __expf(row[m] - mx); row[m] = e; sum += e; }
    float inv = 1.0f / sum;
    for (int m = 0; m < NTOK; ++m) row[m] *= inv;
}

// --------------------------- attn @ v + v_local -----------------------------
__global__ void k_attnout(const float* __restrict__ att, const float* __restrict__ v,
                          const float* __restrict__ vloc, float* __restrict__ o7) {
    int i = blockIdx.x * blockDim.x + threadIdx.x;       // (b,c,n)
    if (i >= BATCH * CDIM * NTOK) return;
    int n = i % NTOK;
    int c = (i / NTOK) % CDIM;
    int b = i / (NTOK * CDIM);
    int h = c >> 6;
    const float* arow = att + ((size_t)(b * HEADS + h) * NTOK + n) * NTOK;
    const float* vrow = v + ((size_t)b * CDIM + c) * NTOK;
    float s = 0.0f;
    for (int m = 0; m < NTOK; ++m) s += arow[m] * vrow[m];
    o7[i] = s + vloc[i];
}

// --------- bilinear 7->56 upsample + hardswish -> f16 token-major -----------
__global__ void k_upsample(const float* __restrict__ o7, _Float16* __restrict__ o56h) {
    size_t i = (size_t)blockIdx.x * blockDim.x + threadIdx.x;
    if (i >= (size_t)BATCH * CDIM * HW) return;
    int p = (int)(i % HW);
    int c = (int)((i / HW) % CDIM);
    int b = (int)(i / ((size_t)HW * CDIM));
    int y = p / RES, x = p % RES;
    float sy = (y + 0.5f) * 0.125f - 0.5f;
    float sx = (x + 0.5f) * 0.125f - 0.5f;
    int y0 = (int)floorf(sy), x0 = (int)floorf(sx);
    float fy = sy - y0, fx = sx - x0;
    int y0c = min(max(y0, 0), HW2 - 1), y1c = min(max(y0 + 1, 0), HW2 - 1);
    int x0c = min(max(x0, 0), HW2 - 1), x1c = min(max(x0 + 1, 0), HW2 - 1);
    const float* pl = o7 + ((size_t)b * CDIM + c) * NTOK;
    float v00 = pl[y0c * HW2 + x0c], v01 = pl[y0c * HW2 + x1c];
    float v10 = pl[y1c * HW2 + x0c], v11 = pl[y1c * HW2 + x1c];
    float v = (1.0f - fy) * ((1.0f - fx) * v00 + fx * v01) +
              fy * ((1.0f - fx) * v10 + fx * v11);
    o56h[((size_t)b * HW + p) * CDIM + c] = (_Float16)hswish(v);
}

// ----------------------------- GroupNorm stats ------------------------------
__global__ void k_gnstats(const float* __restrict__ buf, float* __restrict__ stats,
                          int G, int Cg) {
    int bg = blockIdx.x;
    size_t base = (size_t)bg * Cg * HW;     // channels contiguous per group
    int nelem = Cg * HW;
    float s = 0.0f, s2 = 0.0f;
    for (int i = threadIdx.x; i < nelem; i += blockDim.x) {
        float v = buf[base + i];
        s += v; s2 += v * v;
    }
    __shared__ float sh[256], sh2[256];
    sh[threadIdx.x] = s; sh2[threadIdx.x] = s2;
    __syncthreads();
    for (int off = 128; off > 0; off >>= 1) {
        if ((int)threadIdx.x < off) { sh[threadIdx.x] += sh[threadIdx.x + off];
                                      sh2[threadIdx.x] += sh2[threadIdx.x + off]; }
        __syncthreads();
    }
    if (threadIdx.x == 0) {
        float mu = sh[0] / nelem;
        stats[2 * bg] = mu;
        stats[2 * bg + 1] = sh2[0] / nelem - mu * mu;
    }
}

// --------------- residual 1: x1 = xl*rs1 + GN(a); pack f16*ls2 --------------
__global__ void k_res1(const float* __restrict__ xl, const float* __restrict__ aout,
                       const float* __restrict__ stats, const float* __restrict__ rs1,
                       const float* __restrict__ n1g, const float* __restrict__ n1b,
                       const float* __restrict__ ls2, float* __restrict__ x1,
                       _Float16* __restrict__ x1h) {
    size_t i = (size_t)blockIdx.x * blockDim.x + threadIdx.x;
    if (i >= (size_t)BATCH * CDIM * HW) return;
    int p = (int)(i % HW);
    int c = (int)((i / HW) % CDIM);
    int b = (int)(i / ((size_t)HW * CDIM));
    int gi = b * 32 + (c >> 4);             // 16 channels per group
    float mu = stats[2 * gi], var = stats[2 * gi + 1];
    float gn = (aout[i] - mu) * rsqrtf(var + GEPS) * n1g[c] + n1b[c];
    float v = xl[i] * rs1[c] + gn;
    x1[i] = v;
    x1h[((size_t)b * HW + p) * CDIM + c] = (_Float16)(v * ls2[c]);
}

// ---------------- FFN: GN + hardswish in-place on pw1 output ----------------
__global__ void k_gn_hswish(float* __restrict__ h1, const float* __restrict__ stats,
                            const float* __restrict__ gg, const float* __restrict__ gb) {
    size_t i = (size_t)blockIdx.x * blockDim.x + threadIdx.x;
    if (i >= (size_t)BATCH * HIDC * HW) return;
    int c = (int)((i / HW) % HIDC);
    int b = (int)(i / ((size_t)HW * HIDC));
    int gi = b * 32 + (c >> 6);             // 64 channels per group
    float mu = stats[2 * gi], var = stats[2 * gi + 1];
    float gn = (h1[i] - mu) * rsqrtf(var + GEPS) * gg[c] + gb[c];
    h1[i] = hswish(gn);
}

// ------- hB = hardswish(hA + GN(d)) -> f16 token-major for pw2 GEMM ---------
__global__ void k_hb(const float* __restrict__ hA, const float* __restrict__ d,
                     const float* __restrict__ stats, const float* __restrict__ gg,
                     const float* __restrict__ gb, _Float16* __restrict__ hbh) {
    size_t i = (size_t)blockIdx.x * blockDim.x + threadIdx.x;
    if (i >= (size_t)BATCH * HIDC * HW) return;
    int p = (int)(i % HW);
    int c = (int)((i / HW) % HIDC);
    int b = (int)(i / ((size_t)HW * HIDC));
    int gi = b * 32 + (c >> 6);
    float mu = stats[2 * gi], var = stats[2 * gi + 1];
    float gn = (d[i] - mu) * rsqrtf(var + GEPS) * gg[c] + gb[c];
    float v = hswish(hA[i] + gn);
    hbh[((size_t)b * HW + p) * HIDC + c] = (_Float16)v;
}

// --------------- residual 2: out = x1*rs2 + GN(f2) --------------------------
__global__ void k_res2(const float* __restrict__ x1, const float* __restrict__ f2,
                       const float* __restrict__ stats, const float* __restrict__ rs2,
                       const float* __restrict__ n2g, const float* __restrict__ n2b,
                       float* __restrict__ out) {
    size_t i = (size_t)blockIdx.x * blockDim.x + threadIdx.x;
    if (i >= (size_t)BATCH * CDIM * HW) return;
    int c = (int)((i / HW) % CDIM);
    int b = (int)(i / ((size_t)HW * CDIM));
    int gi = b * 32 + (c >> 4);
    float mu = stats[2 * gi], var = stats[2 * gi + 1];
    float gn = (f2[i] - mu) * rsqrtf(var + GEPS) * n2g[c] + n2b[c];
    out[i] = x1[i] * rs2[c] + gn;
}

// ===========================================================================
extern "C" void kernel_launch(void* const* d_in, const int* in_sizes, int n_in,
                              void* d_out, int out_size, void* d_ws, size_t ws_size,
                              hipStream_t stream) {
    // ---- inputs (setup_inputs dict order, BN tuples expanded g,b,m,v) ----
    const float* X     = (const float*)d_in[0];
    const float* lpu_w = (const float*)d_in[1];
    const float* rs1   = (const float*)d_in[2];
    const float* rs2   = (const float*)d_in[3];
    const float* ls1   = (const float*)d_in[4];
    const float* ls2   = (const float*)d_in[5];
    const float* sc_w  = (const float*)d_in[6];
    const float* sc_b  = (const float*)d_in[7];
    const float *sc_g = (const float*)d_in[8],  *sc_be = (const float*)d_in[9],
                *sc_m = (const float*)d_in[10], *sc_v  = (const float*)d_in[11];
    const float* q_w   = (const float*)d_in[12];
    const float *q_g = (const float*)d_in[13], *q_be = (const float*)d_in[14],
                *q_m = (const float*)d_in[15], *q_v  = (const float*)d_in[16];
    const float* k_w   = (const float*)d_in[17];
    const float *k_g = (const float*)d_in[18], *k_be = (const float*)d_in[19],
                *k_m = (const float*)d_in[20], *k_v  = (const float*)d_in[21];
    const float* v_w   = (const float*)d_in[22];
    const float *v_g = (const float*)d_in[23], *v_be = (const float*)d_in[24],
                *v_m = (const float*)d_in[25], *v_v  = (const float*)d_in[26];
    const float* loc_w = (const float*)d_in[27];
    const float* loc_b = (const float*)d_in[28];
    const float *l_g = (const float*)d_in[29], *l_be = (const float*)d_in[30],
                *l_m = (const float*)d_in[31], *l_v  = (const float*)d_in[32];
    const float* lscale = (const float*)d_in[33];
    const float* abias  = (const float*)d_in[34];
    const float* th1    = (const float*)d_in[35];
    const float* th2    = (const float*)d_in[36];
    const float* out_w  = (const float*)d_in[37];
    const float *o_g = (const float*)d_in[38], *o_be = (const float*)d_in[39],
                *o_m = (const float*)d_in[40], *o_v  = (const float*)d_in[41];
    const float *n1g = (const float*)d_in[42], *n1b = (const float*)d_in[43];
    const float *n2g = (const float*)d_in[44], *n2b = (const float*)d_in[45];
    const float* pw1_w = (const float*)d_in[46];
    const float *g1g = (const float*)d_in[47], *g1b = (const float*)d_in[48];
    const float* dw_w  = (const float*)d_in[49];
    const float *gdg = (const float*)d_in[50], *gdb = (const float*)d_in[51];
    const float* pw2_w = (const float*)d_in[52];
    const float *g2g = (const float*)d_in[53], *g2b = (const float*)d_in[54];
    const int*   idxs  = (const int*)d_in[n_in - 1];

    // ---- workspace carve (256-B aligned) ----
    char* ws = (char*)d_ws;
    size_t off = 0;
    auto carve = [&](size_t bytes) -> void* {
        void* p = ws + off;
        off += (bytes + 255) & ~(size_t)255;
        return p;
    };
    const size_t NE  = (size_t)BATCH * CDIM * HW;    // 12,845,056
    const size_t NEH = (size_t)BATCH * HIDC * HW;    // 51,380,224
    const size_t NT  = (size_t)BATCH * CDIM * NTOK;  // 200,704

    float*    XL    = (float*)carve(NE * 4);
    _Float16* XSCH  = (_Float16*)carve(NT * 2);
    float*    QB    = (float*)carve(NT * 4);
    float*    KB    = (float*)carve(NT * 4);
    float*    VB    = (float*)carve(NT * 4);
    float*    VLOC  = (float*)carve(NT * 4);
    float*    QN    = (float*)carve((size_t)BATCH * HEADS * NTOK * 4);
    float*    KN    = (float*)carve((size_t)BATCH * CDIM * 4);
    float*    ATT0  = (float*)carve((size_t)BATCH * HEADS * NTOK * NTOK * 4);
    float*    ATT1  = (float*)carve((size_t)BATCH * HEADS * NTOK * NTOK * 4);
    float*    O7    = (float*)carve(NT * 4);
    _Float16* O56H  = (_Float16*)carve(NE * 2);
    float*    AOUT  = (float*)carve(NE * 4);
    float*    STATS = (float*)carve((size_t)BATCH * 32 * 2 * 4);
    float*    X1    = (float*)carve(NE * 4);
    _Float16* X1H   = (_Float16*)carve(NE * 2);
    float*    H1    = (float*)carve(NEH * 4);
    float*    DBUF  = (float*)carve(NEH * 4);
    _Float16* HBH   = (_Float16*)carve(NEH * 2);
    float*    F2    = (float*)carve(NE * 4);
    _Float16* WQH   = (_Float16*)carve((size_t)CDIM * CDIM * 2);
    _Float16* WKH   = (_Float16*)carve((size_t)CDIM * CDIM * 2);
    _Float16* WVH   = (_Float16*)carve((size_t)CDIM * CDIM * 2);
    _Float16* WOH   = (_Float16*)carve((size_t)CDIM * CDIM * 2);
    _Float16* WP1H  = (_Float16*)carve((size_t)HIDC * CDIM * 2);
    _Float16* WP2H  = (_Float16*)carve((size_t)CDIM * HIDC * 2);
    (void)ws_size; (void)in_sizes; (void)out_size;

    const int TPB = 256;
    auto blocks = [](size_t n, int t) { return (unsigned)((n + t - 1) / t); };
    auto gemm = [&](const _Float16* Wf, const _Float16* Bt, float* out,
                    int M, int K, int P, const float* g, const float* be,
                    const float* mm, const float* vv) {
        int tiles = (M >> 6) * ((P + 31) >> 5);     // 64x32 tiles per wave
        dim3 grid((tiles + 3) / 4, BATCH);
        k_gemm_bn<<<grid, 128, 0, stream>>>(Wf, Bt, out, M, K, P, g, be, mm, vv);
    };

    // ---- weight conversions to f16 ----
    k_cvt<<<blocks(CDIM * CDIM, TPB), TPB, 0, stream>>>(q_w, WQH, (size_t)CDIM * CDIM);
    k_cvt<<<blocks(CDIM * CDIM, TPB), TPB, 0, stream>>>(k_w, WKH, (size_t)CDIM * CDIM);
    k_cvt<<<blocks(CDIM * CDIM, TPB), TPB, 0, stream>>>(v_w, WVH, (size_t)CDIM * CDIM);
    k_cvt<<<blocks(CDIM * CDIM, TPB), TPB, 0, stream>>>(out_w, WOH, (size_t)CDIM * CDIM);
    k_cvt<<<blocks((size_t)HIDC * CDIM, TPB), TPB, 0, stream>>>(pw1_w, WP1H, (size_t)HIDC * CDIM);
    k_cvt<<<blocks((size_t)CDIM * HIDC, TPB), TPB, 0, stream>>>(pw2_w, WP2H, (size_t)CDIM * HIDC);

    // ---- 1) LPU: xl = x + dwconv(x) ----
    k_dwconv<<<blocks(NE, TPB), TPB, 0, stream>>>(X, lpu_w, XL, CDIM, 1, NE);

    // ---- 2) stride conv + BN -> f16 tokens (ls1 fused) ----
    k_stride_bn<<<blocks(NT, TPB), TPB, 0, stream>>>(XL, sc_w, sc_b, ls1,
                                                     sc_g, sc_be, sc_m, sc_v, XSCH);

    // ---- 3) q/k/v 1x1 convs as WMMA GEMM + BN epilogue ----
    gemm(WQH, XSCH, QB, CDIM, CDIM, NTOK, q_g, q_be, q_m, q_v);
    gemm(WKH, XSCH, KB, CDIM, CDIM, NTOK, k_g, k_be, k_m, k_v);
    gemm(WVH, XSCH, VB, CDIM, CDIM, NTOK, v_g, v_be, v_m, v_v);

    // ---- 4) v_local grouped 3x3 conv + BN ----
    k_vlocal<<<blocks(NT, TPB), TPB, 0, stream>>>(VB, loc_w, loc_b, l_g, l_be, l_m, l_v, VLOC);

    // ---- 5) attention ----
    k_qnorm<<<blocks((size_t)BATCH * HEADS * NTOK, TPB), TPB, 0, stream>>>(QB, QN);
    k_knorm<<<blocks((size_t)BATCH * CDIM, TPB), TPB, 0, stream>>>(KB, KN);
    size_t natt = (size_t)BATCH * HEADS * NTOK * NTOK;
    k_logits<<<blocks(natt, TPB), TPB, 0, stream>>>(QB, KB, QN, KN, lscale, abias, idxs, ATT0);
    k_thmix<<<blocks(natt, TPB), TPB, 0, stream>>>(ATT0, th1, ATT1);
    k_softmax<<<blocks((size_t)BATCH * HEADS * NTOK, TPB), TPB, 0, stream>>>(ATT1);
    k_thmix<<<blocks(natt, TPB), TPB, 0, stream>>>(ATT1, th2, ATT0);
    k_attnout<<<blocks(NT, TPB), TPB, 0, stream>>>(ATT0, VB, VLOC, O7);

    // ---- 6) bilinear 7->56 + hardswish -> f16 tokens ----
    k_upsample<<<blocks(NE, TPB), TPB, 0, stream>>>(O7, O56H);

    // ---- 7) out 1x1 conv (WMMA) + BN ----
    gemm(WOH, O56H, AOUT, CDIM, CDIM, HW, o_g, o_be, o_m, o_v);

    // ---- 8) residual 1 with GN; pack FFN input (ls2 fused) ----
    k_gnstats<<<BATCH * 32, 256, 0, stream>>>(AOUT, STATS, 32, CDIM / 32);
    k_res1<<<blocks(NE, TPB), TPB, 0, stream>>>(XL, AOUT, STATS, rs1, n1g, n1b, ls2, X1, X1H);

    // ---- 9) FFN pw1 (WMMA), GN + hardswish ----
    gemm(WP1H, X1H, H1, HIDC, CDIM, HW, nullptr, nullptr, nullptr, nullptr);
    k_gnstats<<<BATCH * 32, 256, 0, stream>>>(H1, STATS, 32, HIDC / 32);
    k_gn_hswish<<<blocks(NEH, TPB), TPB, 0, stream>>>(H1, STATS, g1g, g1b);

    // ---- 10) depthwise 3x3, GN, add + hardswish -> f16 tokens ----
    k_dwconv<<<blocks(NEH, TPB), TPB, 0, stream>>>(H1, dw_w, DBUF, HIDC, 0, NEH);
    k_gnstats<<<BATCH * 32, 256, 0, stream>>>(DBUF, STATS, 32, HIDC / 32);
    k_hb<<<blocks(NEH, TPB), TPB, 0, stream>>>(H1, DBUF, STATS, gdg, gdb, HBH);

    // ---- 11) pw2 (WMMA), GN, residual 2 -> d_out ----
    gemm(WP2H, HBH, F2, CDIM, HIDC, HW, nullptr, nullptr, nullptr, nullptr);
    k_gnstats<<<BATCH * 32, 256, 0, stream>>>(F2, STATS, 32, CDIM / 32);
    k_res2<<<blocks(NE, TPB), TPB, 0, stream>>>(X1, F2, STATS, rs2, n2g, n2b, (float*)d_out);
}
